// CapsNet_Recon_FLOWER102_12601434046790
// MI455X (gfx1250) — compile-verified, matrix operand order checked by reading, not run on hardware
//
#include <hip/hip_runtime.h>
#include <cstdint>

// ---------------------------------------------------------------------------
// CDNA5 (gfx1250) CapsNet forward: bf16 WMMA implicit-GEMM convs + fc3,
// VALU routing. Wave32 only. Multi-N-tile waves amortize the im2col gather.
// ---------------------------------------------------------------------------

typedef __attribute__((ext_vector_type(16))) __bf16 v16bf;
typedef __attribute__((ext_vector_type(8)))  float  v8f;

union AFragU {
  v16bf v;
  unsigned short us[16];
  uint4 q[2];
};

__device__ __forceinline__ unsigned short f2bf(float f) {
  // round-to-nearest-even f32 -> bf16
  unsigned int u = __builtin_bit_cast(unsigned int, f);
  unsigned int r = (u + 0x7FFFu + ((u >> 16) & 1u)) >> 16;
  return (unsigned short)r;
}

// ------------------------- elementwise converters --------------------------

__global__ void k_cvt_bf16(const float* __restrict__ in,
                           unsigned short* __restrict__ out, int n) {
  int i = blockIdx.x * blockDim.x + threadIdx.x;
  if (i < n) out[i] = f2bf(in[i]);
}

__global__ void k_cvt_pad_rows(const float* __restrict__ in,
                               unsigned short* __restrict__ out,
                               int K, int KPAD, int total) {
  int i = blockIdx.x * blockDim.x + threadIdx.x;
  if (i >= total) return;
  int row = i / KPAD;
  int k = i - row * KPAD;
  out[i] = (k < K) ? f2bf(in[(size_t)row * K + k]) : (unsigned short)0;
}

__global__ void k_zero_f32(float* __restrict__ p, int n) {
  int i = blockIdx.x * blockDim.x + threadIdx.x;
  if (i < n) p[i] = 0.f;
}

// --------------------- implicit im2col conv as WMMA GEMM -------------------
// A: bf16 activations [B][CIN][H][Wd], implicit im2col -> [M][K]
// Bw: bf16 weights [NOUT][KPAD] (k-contiguous rows, zero padded to KPAD)
// out: [B][NOUT][OH][OW], f32 or bf16, + bias, optional ReLU.
// One wave = 16 x (16*NWN) output tile: one A fragment feeds NWN WMMAs.
// Waves in a block share the M-tile -> identical A gathers hit WGP$.

template <int CIN, int KH, int KW, int H, int Wd, int OH, int OW, int STRIDE,
          int K, int KPAD, int NOUT, int NWN, bool RELU, bool OUT_BF16>
__global__ void k_conv_gemm_wmma(const unsigned short* __restrict__ A,
                                 const unsigned short* __restrict__ Bw,
                                 const float* __restrict__ bias,
                                 void* __restrict__ outp, int M) {
  const int MT = M / 16;
  const int NTG = NOUT / (16 * NWN);  // n-tile groups
  int wave = blockIdx.x * (blockDim.x >> 5) + (threadIdx.x >> 5);
  if (wave >= MT * NTG) return;
  int mt = wave / NTG;      // consecutive waves share mt -> shared A gathers
  int ntg = wave - mt * NTG;
  int lane = threadIdx.x & 31;
  int half = lane >> 4;     // 0: lanes 0-15, 1: lanes 16-31
  int l16 = lane & 15;

  // A-fragment row owned by this lane
  int m = mt * 16 + l16;
  int b = m / (OH * OW);
  int rem = m - b * (OH * OW);
  int oy = rem / OW;
  int ox = rem - oy * OW;
  const unsigned short* Abase =
      A + (size_t)b * CIN * H * Wd + (size_t)(oy * STRIDE) * Wd + ox * STRIDE;

  // B-fragment columns owned by this lane (weights k-contiguous)
  const unsigned short* Brow[NWN];
#pragma unroll
  for (int s = 0; s < NWN; ++s) {
    int n = (ntg * NWN + s) * 16 + l16;
    Brow[s] = Bw + (size_t)n * KPAD + half * 8;
  }

  v8f acc[NWN];
#pragma unroll
  for (int s = 0; s < NWN; ++s)
    acc[s] = (v8f){0.f, 0.f, 0.f, 0.f, 0.f, 0.f, 0.f, 0.f};

  for (int kc = 0; kc < KPAD / 32; ++kc) {
    int kbase = kc * 32;
    // prefetch weights 4 chunks (128 k) ahead (global_prefetch_b8; speculative, droppable)
#pragma unroll
    for (int s = 0; s < NWN; ++s)
      __builtin_prefetch(Brow[s] + kbase + 128, 0, 1);

    // A fragment: gather 16 bf16
    // (ISA 16-bit A layout: e -> K = (e/8)*16 + half*8 + e%8)
    AFragU af;
#pragma unroll
    for (int e = 0; e < 16; ++e) {
      int kk = kbase + ((e >> 3) << 4) + (half << 3) + (e & 7);
      unsigned short val = 0;
      if ((KPAD == K) || (kk < K)) {
        int ic = kk / (KH * KW);
        int r2 = kk - ic * (KH * KW);
        int ky = r2 / KW;
        int kx = r2 - ky * KW;
        val = Abase[(size_t)ic * (H * Wd) + ky * Wd + kx];
      }
      af.us[e] = val;
    }
    // B fragments: aligned 16B loads; one A frag drives NWN WMMAs
#pragma unroll
    for (int s = 0; s < NWN; ++s) {
      AFragU bfrag;
      bfrag.q[0] = *(const uint4*)(Brow[s] + kbase);
      bfrag.q[1] = *(const uint4*)(Brow[s] + kbase + 16);
      acc[s] = __builtin_amdgcn_wmma_f32_16x16x32_bf16(
          false, af.v, false, bfrag.v, (short)0, acc[s], false, false);
    }
  }

#pragma unroll
  for (int s = 0; s < NWN; ++s) {
    int n = (ntg * NWN + s) * 16 + l16;
    float bs = bias[n];
#pragma unroll
    for (int r = 0; r < 8; ++r) {
      int mr = mt * 16 + r + half * 8;  // C/D layout: VGPR r + lane-half M block
      float val = acc[s][r] + bs;
      if (RELU) val = val > 0.f ? val : 0.f;
      int b2 = mr / (OH * OW);
      int rm = mr - b2 * (OH * OW);
      int oy2 = rm / OW;
      int ox2 = rm - oy2 * OW;
      size_t oidx = (((size_t)b2 * NOUT + n) * OH + oy2) * OW + ox2;
      if (OUT_BF16)
        ((unsigned short*)outp)[oidx] = f2bf(val);
      else
        ((float*)outp)[oidx] = val;
    }
  }
}

// ------------------------------ squash(u) ----------------------------------
__global__ void k_squash8(const float* __restrict__ pc, float* __restrict__ u,
                          int ncaps) {
  int i = blockIdx.x * blockDim.x + threadIdx.x;
  if (i >= ncaps) return;
  const float* p = pc + (size_t)i * 8;
  float s[8];
  float sq = 0.f;
#pragma unroll
  for (int d = 0; d < 8; ++d) {
    s[d] = p[d];
    sq += s[d] * s[d];
  }
  float scale = (sq / (1.f + sq)) * rsqrtf(sq + 1e-8f);
  float* o = u + (size_t)i * 8;
#pragma unroll
  for (int d = 0; d < 8; ++d) o[d] = s[d] * scale;
}

// ------------------- routing: softmax stats over class axis ----------------
__global__ void k_softmax_stats(const float* __restrict__ blog,
                                float2* __restrict__ stats, int BN) {
  int i = blockIdx.x * blockDim.x + threadIdx.x;
  if (i >= BN) return;
  int b = i / 2592;
  int n = i - b * 2592;
  const float* p = blog + (size_t)b * 102 * 2592 + n;
  float mx = -1e30f;
  for (int o = 0; o < 102; ++o) {
    float v = p[(size_t)o * 2592];
    mx = v > mx ? v : mx;
  }
  float sum = 0.f;
  for (int o = 0; o < 102; ++o) sum += expf(p[(size_t)o * 2592] - mx);
  stats[i] = make_float2(mx, 1.f / sum);
}

// -------- routing: s = sum_n c * xhat (xhat recomputed), v = squash(s) -----
// one 256-thread block per (b,o); W[o] slices (102 x 1.33MB) live in L2.
__global__ void k_route_sv(const float* __restrict__ blog,
                           const float2* __restrict__ stats,
                           const float* __restrict__ u,
                           const float* __restrict__ W,
                           float* __restrict__ vout) {
  int bo = blockIdx.x;           // b*102 + o
  int b = bo / 102;
  int o = bo - b * 102;
  int tid = threadIdx.x;
  int lane = tid & 31;
  int wid = tid >> 5;

  float acc[16];
#pragma unroll
  for (int d = 0; d < 16; ++d) acc[d] = 0.f;

  const float* blogRow = blog + (size_t)bo * 2592;
  const float* ub = u + (size_t)b * 2592 * 8;
  const float2* st = stats + (size_t)b * 2592;

  for (int n = tid; n < 2592; n += 256) {
    float2 s2 = st[n];
    float cv = expf(blogRow[n] - s2.x) * s2.y;
    float uu[8];
    const float* up = ub + (size_t)n * 8;
#pragma unroll
    for (int i = 0; i < 8; ++i) uu[i] = up[i];
    const float* wp = W + ((size_t)o * 2592 + n) * 128;
#pragma unroll
    for (int d = 0; d < 16; ++d) {
      float xh = 0.f;
#pragma unroll
      for (int i = 0; i < 8; ++i) xh += wp[d * 8 + i] * uu[i];
      acc[d] += cv * xh;
    }
  }

  __shared__ float lds[8 * 16 + 16];
#pragma unroll
  for (int d = 0; d < 16; ++d) {
    float a = acc[d];
    for (int off = 16; off > 0; off >>= 1) a += __shfl_down(a, off, 32);
    if (lane == 0) lds[wid * 16 + d] = a;
  }
  __syncthreads();
  if (tid < 16) {
    float s = 0.f;
    for (int w = 0; w < 8; ++w) s += lds[w * 16 + tid];
    lds[128 + tid] = s;
  }
  __syncthreads();
  if (tid < 16) {
    float sq = 0.f;
    for (int d = 0; d < 16; ++d) {
      float t = lds[128 + d];
      sq += t * t;
    }
    float scale = (sq / (1.f + sq)) * rsqrtf(sq + 1e-8f);
    vout[(size_t)bo * 16 + tid] = lds[128 + tid] * scale;
  }
}

// ----------------- routing: b += <xhat(b,o,n,:), v(b,o,:)> -----------------
__global__ void k_route_updateb(float* __restrict__ blog,
                                const float* __restrict__ u,
                                const float* __restrict__ W,
                                const float* __restrict__ v, int total) {
  int idx = blockIdx.x * blockDim.x + threadIdx.x;
  if (idx >= total) return;
  int n = idx % 2592;
  int bo = idx / 2592;
  int b = bo / 102;
  int o = bo - b * 102;
  const float* up = u + ((size_t)b * 2592 + n) * 8;
  const float* vp = v + (size_t)bo * 16;
  const float* wp = W + ((size_t)o * 2592 + n) * 128;
  float uu[8];
#pragma unroll
  for (int i = 0; i < 8; ++i) uu[i] = up[i];
  float dot = 0.f;
#pragma unroll
  for (int d = 0; d < 16; ++d) {
    float xh = 0.f;
#pragma unroll
    for (int i = 0; i < 8; ++i) xh += wp[d * 8 + i] * uu[i];
    dot += xh * vp[d];
  }
  blog[idx] += dot;
}

// ------------------------------ decoder fc1/fc2 ----------------------------
__global__ void k_fc1_mask(const float* __restrict__ v,
                           const int* __restrict__ targets,
                           const float* __restrict__ w,
                           const float* __restrict__ bias,
                           float* __restrict__ h1) {
  int idx = blockIdx.x * blockDim.x + threadIdx.x;
  if (idx >= 32 * 512) return;
  int b = idx / 512;
  int j = idx - b * 512;
  int t = targets[b];
  const float* vp = v + ((size_t)b * 102 + t) * 16;
  float acc = bias[j];
#pragma unroll
  for (int d = 0; d < 16; ++d)
    acc += vp[d] * w[(size_t)(t * 16 + d) * 512 + j];
  h1[idx] = acc > 0.f ? acc : 0.f;
}

__global__ void k_fc2(const float* __restrict__ h1, const float* __restrict__ w,
                      const float* __restrict__ bias,
                      unsigned short* __restrict__ h2b) {
  int idx = blockIdx.x * blockDim.x + threadIdx.x;
  if (idx >= 32 * 1024) return;
  int b = idx / 1024;
  int j = idx - b * 1024;
  const float* hp = h1 + (size_t)b * 512;
  float acc = bias[j];
  for (int k = 0; k < 512; ++k) acc += hp[k] * w[(size_t)k * 1024 + j];
  acc = acc > 0.f ? acc : 0.f;
  h2b[idx] = f2bf(acc);
}

// ----------------- fc3 (1024 -> 37632) WMMA + bias + sigmoid ---------------
// One wave per N-tile computes BOTH M-tiles (M=32): the strided f32->bf16
// B gather is shared by two WMMAs.
__global__ void k_fc3_wmma(const unsigned short* __restrict__ h2b,
                           const float* __restrict__ w3,
                           const float* __restrict__ b3,
                           float* __restrict__ out) {
  const int NT = 37632 / 16;  // 2352 column tiles
  int wave = blockIdx.x * (blockDim.x >> 5) + (threadIdx.x >> 5);
  if (wave >= NT) return;
  int nt = wave;
  int lane = threadIdx.x & 31;
  int half = lane >> 4;
  int l16 = lane & 15;
  int n = nt * 16 + l16;

  const unsigned short* Arow0 = h2b + (size_t)l16 * 1024 + half * 8;
  const unsigned short* Arow1 = h2b + (size_t)(16 + l16) * 1024 + half * 8;
  const float* Bcol = w3 + n;

  v8f c0 = {0.f, 0.f, 0.f, 0.f, 0.f, 0.f, 0.f, 0.f};
  v8f c1 = {0.f, 0.f, 0.f, 0.f, 0.f, 0.f, 0.f, 0.f};
  for (int kc = 0; kc < 32; ++kc) {
    int kbase = kc * 32;
    AFragU af0, af1;
    af0.q[0] = *(const uint4*)(Arow0 + kbase);
    af0.q[1] = *(const uint4*)(Arow0 + kbase + 16);
    af1.q[0] = *(const uint4*)(Arow1 + kbase);
    af1.q[1] = *(const uint4*)(Arow1 + kbase + 16);
    AFragU bfrag;
#pragma unroll
    for (int e = 0; e < 16; ++e) {
      int kk = kbase + ((e >> 3) << 4) + (half << 3) + (e & 7);
      bfrag.us[e] = f2bf(Bcol[(size_t)kk * 37632]);
    }
    c0 = __builtin_amdgcn_wmma_f32_16x16x32_bf16(false, af0.v, false, bfrag.v,
                                                 (short)0, c0, false, false);
    c1 = __builtin_amdgcn_wmma_f32_16x16x32_bf16(false, af1.v, false, bfrag.v,
                                                 (short)0, c1, false, false);
  }
  float bs = b3[n];
#pragma unroll
  for (int r = 0; r < 8; ++r) {
    int mr0 = r + half * 8;
    float v0 = c0[r] + bs;
    v0 = 1.f / (1.f + expf(-v0));
    out[(size_t)mr0 * 37632 + n] = v0;
    int mr1 = 16 + r + half * 8;
    float v1 = c1[r] + bs;
    v1 = 1.f / (1.f + expf(-v1));
    out[(size_t)mr1 * 37632 + n] = v1;
  }
}

// ---------------------------------------------------------------------------
extern "C" void kernel_launch(void* const* d_in, const int* in_sizes, int n_in,
                              void* d_out, int out_size, void* d_ws,
                              size_t ws_size, hipStream_t stream) {
  (void)in_sizes; (void)n_in; (void)out_size; (void)ws_size;

  const float* x     = (const float*)d_in[0];
  const int*   tgt   = (const int*)d_in[1];
  const float* w1    = (const float*)d_in[2];
  const float* b1    = (const float*)d_in[3];
  const float* w2    = (const float*)d_in[4];
  const float* b2    = (const float*)d_in[5];
  const float* Wt    = (const float*)d_in[6];
  const float* fc1w  = (const float*)d_in[7];
  const float* fc1b  = (const float*)d_in[8];
  const float* fc2w  = (const float*)d_in[9];
  const float* fc2b  = (const float*)d_in[10];
  const float* fc3w  = (const float*)d_in[11];
  const float* fc3b  = (const float*)d_in[12];

  // workspace carve-up (256B aligned)
  char* ws = (char*)d_ws;
  size_t off = 0;
  auto carve = [&](size_t bytes) -> void* {
    void* p = ws + off;
    off = (off + bytes + 255) & ~(size_t)255;
    return p;
  };
  unsigned short* xb    = (unsigned short*)carve(1204224u * 2);   // x bf16
  unsigned short* w1b   = (unsigned short*)carve(98304u * 2);     // [256][384] padded
  unsigned short* featb = (unsigned short*)carve(9469952u * 2);   // conv1 out bf16
  unsigned short* w2b   = (unsigned short*)carve(5308416u * 2);   // [256][20736]
  float*  pcf   = (float*)carve(663552u * 4);                     // primary caps raw
  float*  uf    = (float*)carve(663552u * 4);                     // squashed u
  float*  blog  = (float*)carve(8515584u * 4);                    // routing logits
  float2* stats = (float2*)carve(82944u * 8);                     // softmax stats
  float*  h1    = (float*)carve(16384u * 4);
  unsigned short* h2b = (unsigned short*)carve(32768u * 2);

  float* vout  = (float*)d_out;            // [32][102][16]
  float* recon = (float*)d_out + 52224;    // [32][3][112][112]

  // 1. convert activations/weights to bf16
  k_cvt_bf16<<<4704, 256, 0, stream>>>(x, xb, 1204224);
  k_cvt_pad_rows<<<384, 256, 0, stream>>>(w1, w1b, 363, 384, 98304);
  k_cvt_bf16<<<20736, 256, 0, stream>>>(w2, w2b, 5308416);

  // 2. conv1 (implicit GEMM, bf16 WMMA, 4 N-tiles/wave, ReLU, bf16 out)
  //    waves = 2312 * 4 = 9248 -> 1156 blocks
  k_conv_gemm_wmma<3, 11, 11, 112, 112, 34, 34, 3, 363, 384, 256, 4, true, true>
      <<<1156, 256, 0, stream>>>(xb, w1b, b1, (void*)featb, 36992);

  // 3. primary-caps conv (K=20736, 2 N-tiles/wave, f32 out)
  //    waves = 162 * 8 = 1296 -> 162 blocks; 648 chunks x 2 wmma each
  k_conv_gemm_wmma<256, 9, 9, 34, 34, 9, 9, 3, 20736, 20736, 256, 2, false,
                   false>
      <<<162, 256, 0, stream>>>(featb, w2b, b2, (void*)pcf, 2592);

  // 4. squash primary capsules
  k_squash8<<<324, 256, 0, stream>>>(pcf, uf, 82944);

  // 5. dynamic routing (3 iterations)
  k_zero_f32<<<33264, 256, 0, stream>>>(blog, 8515584);
  for (int it = 0; it < 3; ++it) {
    k_softmax_stats<<<324, 256, 0, stream>>>(blog, stats, 82944);
    k_route_sv<<<3264, 256, 0, stream>>>(blog, stats, uf, Wt, vout);
    if (it < 2)
      k_route_updateb<<<33264, 256, 0, stream>>>(blog, uf, Wt, vout, 8515584);
  }

  // 6. decoder
  k_fc1_mask<<<64, 256, 0, stream>>>(vout, tgt, fc1w, fc1b, h1);
  k_fc2<<<128, 256, 0, stream>>>(h1, fc2w, fc2b, h2b);
  k_fc3_wmma<<<294, 256, 0, stream>>>(h2b, fc3w, fc3b, recon);
}
